// RNNModel_21801253994762
// MI455X (gfx1250) — matrix operand order repaired and checked by reading, not compiled, not run on hardware
//
#include <hip/hip_runtime.h>

// ---------------------------------------------------------------------------
// 2-layer LSTM LM on gfx1250 (MI455X), bf16 WMMA + f32 accum.
//   V=80, D_IN=8, H=256, B=4096, T=80
// Strategy:
//   * prepass: G0p[v][p] = (emb @ W_ih0^T) with gate-permuted columns (f32)
//              Wp = gate-permuted bf16 copies of W_hh0, W_ih1, W_hh1
//              fcp = bf16 fc_W padded to 96 rows
//   * main: 64 workgroups x 512 threads (16 waves). Each WG owns 64 batch
//     rows and loops t=0..79. Wave w owns permuted gate columns
//     [64w, 64w+64) == hidden cols j in [16w,16w+16) with all 4 gates as
//     4 adjacent 16-wide N-tiles -> pointwise LSTM update is lane-local,
//     cell state c stays in registers for all 80 steps.
//     h (bf16) lives in LDS as the WMMA A operand; weights stream from L2
//     every step. A laundered per-lane *offset* (not pointer) keeps the
//     loads un-hoistable across the time loop while preserving the global
//     address space -> global_load_b128 saddr-form, LOADcnt only.
// ---------------------------------------------------------------------------

#define VOCAB   80
#define D_IN    8
#define HDIM    256
#define BATCH   4096
#define TSTEPS  80
#define MTILE   64      // batch rows per workgroup
#define NWAVES  16

typedef __attribute__((ext_vector_type(16))) __bf16 v16bf;
typedef __attribute__((ext_vector_type(8)))  float  v8f;

// gate-permuted row index: p = 64*(j>>4) + 16*g + (j&15)  <->  src row 256*g + j
__host__ __device__ inline int perm_src_row(int p) {
    int jhi = p >> 6, g = (p >> 4) & 3, jlo = p & 15;
    return 256 * g + 16 * jhi + jlo;
}

// ---------------- prepass kernels ----------------

__global__ void build_g0(const float* __restrict__ emb,
                         const float* __restrict__ W_ih0,
                         float* __restrict__ G0p) {
    int idx = blockIdx.x * blockDim.x + threadIdx.x;
    if (idx >= VOCAB * 1024) return;
    int v = idx >> 10, p = idx & 1023;
    int row = perm_src_row(p);
    float s = 0.f;
#pragma unroll
    for (int k = 0; k < D_IN; ++k)
        s += emb[v * D_IN + k] * W_ih0[row * D_IN + k];
    G0p[idx] = s;
}

__global__ void permute_w(const float* __restrict__ src,  // [1024][256]
                          __bf16* __restrict__ dst) {     // [1024][256] permuted
    int idx = blockIdx.x * blockDim.x + threadIdx.x;
    if (idx >= 1024 * HDIM) return;
    int p = idx >> 8, k = idx & 255;
    dst[idx] = (__bf16)src[perm_src_row(p) * HDIM + k];
}

__global__ void conv_fc(const float* __restrict__ fcW,    // [80][256]
                        __bf16* __restrict__ dst) {       // [96][256], zero pad
    int idx = blockIdx.x * blockDim.x + threadIdx.x;
    if (idx >= 96 * HDIM) return;
    int row = idx >> 8, k = idx & 255;
    dst[idx] = (row < VOCAB) ? (__bf16)fcW[row * HDIM + k] : (__bf16)0.0f;
}

// ---------------- fragment loaders (contiguous k = 16h + s order, same for A and B) ------

// A 16x32 bf16: lane l -> row l&15; VGPR0-7 = K {k0+16h .. k0+16h+15}
__device__ inline v16bf load_a(const __bf16* hs, int mt, int k0, int l15, int half) {
    union { v16bf v; uint4 q[2]; } u;
    const __bf16* p = hs + (16 * mt + l15) * HDIM + k0 + 16 * half;
    u.q[0] = *(const uint4*)p;
    u.q[1] = *(const uint4*)(p + 8);
    return u.v;
}

// B 32x16 bf16 (B[k][n] = W[n][k], W row-major): lane l -> col l&15;
// VGPR0-7 = K {k0+16h .. k0+16h+15}: contiguous 32 bytes of row n.
__device__ inline v16bf load_b(const __bf16* W, int col, int k0, int half) {
    union { v16bf v; uint4 q[2]; } u;
    const __bf16* p = W + (size_t)col * HDIM + k0 + 16 * half;
    u.q[0] = *(const uint4*)p;
    u.q[1] = *(const uint4*)(p + 8);
    return u.v;
}

// acc[mt][nt] += A(h-tile mt) x B(W cols 64*wv+16*nt ...), K = 256
__device__ inline void mm_acc(const __bf16* hs, const __bf16* __restrict__ W,
                              int wv, int l15, int half, v8f acc[4][4]) {
    // Launder the per-lane element offset (NOT the pointer): the weight loads
    // become un-hoistable across the 80-step time loop (prevents the B-frag
    // spill-to-scratch), while W keeps its global address-space provenance so
    // the loads lower to global_load_b128 (saddr + voffset), not flat_load
    // (flat would also burn DScnt and the LDS port we need for A-fragments).
    int base_off = (64 * wv + l15) * HDIM + 16 * half;
    asm volatile("" : "+v"(base_off));
    const __bf16* wrow = W + base_off;
#pragma clang loop unroll_count(2)
    for (int ks = 0; ks < 8; ++ks) {
        const int k0 = 32 * ks;
        // global weight loads first (longest latency)
        v16bf b[4];
#pragma unroll
        for (int nt = 0; nt < 4; ++nt) {
            union { v16bf v; uint4 q[2]; } u;
            const __bf16* p = wrow + 16 * nt * HDIM + k0;
            u.q[0] = *(const uint4*)p;
            u.q[1] = *(const uint4*)(p + 8);
            b[nt] = u.v;
        }
        // LDS activation loads
        v16bf a[4];
#pragma unroll
        for (int mt = 0; mt < 4; ++mt) a[mt] = load_a(hs, mt, k0, l15, half);
#pragma unroll
        for (int nt = 0; nt < 4; ++nt)
#pragma unroll
            for (int mt = 0; mt < 4; ++mt)
                acc[mt][nt] = __builtin_amdgcn_wmma_f32_16x16x32_bf16(
                    false, a[mt], false, b[nt], (short)0, acc[mt][nt], false, false);
    }
}

__device__ inline float fsig(float x)  { return 1.f / (1.f + __expf(-x)); }
__device__ inline float ftanh(float x) { return 1.f - 2.f / (1.f + __expf(2.f * x)); }

// lane-local pointwise LSTM update; nt 0..3 hold gates i,f,g,o for the same j
__device__ inline void lstm_update(v8f acc[4][4], float cr[4][8], __bf16* hs,
                                   const int* x /*null for layer2*/,
                                   const float* __restrict__ G0p, int t,
                                   int bbase, int wv, int l15, int half,
                                   bool last, float* hnOut, float* cnOut) {
    const int j = 16 * wv + l15;
#pragma unroll
    for (int mt = 0; mt < 4; ++mt) {
#pragma unroll
        for (int r = 0; r < 8; ++r) {
            const int m = 16 * mt + 8 * half + r;
            float iv = acc[mt][0][r], fv = acc[mt][1][r];
            float gv = acc[mt][2][r], ov = acc[mt][3][r];
            if (x) {  // layer-1: add precomputed token gate row
                int tok = x[(bbase + m) * TSTEPS + t];
                const float* g0 = G0p + tok * 1024 + 64 * wv + l15;
                iv += g0[0]; fv += g0[16]; gv += g0[32]; ov += g0[48];
            }
            float cnew = fsig(fv) * cr[mt][r] + fsig(iv) * ftanh(gv);
            float h    = fsig(ov) * ftanh(cnew);
            cr[mt][r]  = cnew;
            hs[m * HDIM + j] = (__bf16)h;
            if (last) {
                hnOut[(size_t)(bbase + m) * HDIM + j] = h;
                cnOut[(size_t)(bbase + m) * HDIM + j] = cnew;
            }
        }
    }
}

// ---------------- main persistent kernel ----------------

__global__ __launch_bounds__(512, 1)
void lstm_kernel(const int*   __restrict__ x,
                 const float* __restrict__ h0,
                 const float* __restrict__ c0,
                 const float* __restrict__ G0p,
                 const __bf16* __restrict__ Wp0,   // layer1 recurrent (perm)
                 const __bf16* __restrict__ Wpi1,  // layer2 input    (perm)
                 const __bf16* __restrict__ Wp1,   // layer2 recurrent(perm)
                 const __bf16* __restrict__ fcp,   // [96][256] bf16
                 const float* __restrict__ fc_b,
                 float* __restrict__ out) {
    __shared__ __align__(16) __bf16 h1s[MTILE * HDIM];
    __shared__ __align__(16) __bf16 h2s[MTILE * HDIM];

    const int lane = threadIdx.x & 31;
    const int wv   = threadIdx.x >> 5;
    const int l15  = lane & 15;
    const int half = lane >> 4;
    const int bbase = blockIdx.x * MTILE;

    float* logits = out;
    float* hn     = out + (size_t)BATCH * VOCAB;
    float* cn     = hn + (size_t)2 * BATCH * HDIM;

    // init state (honor h0/c0 inputs)
    float c1r[4][8], c2r[4][8];
    {
        const int j = 16 * wv + l15;
#pragma unroll
        for (int mt = 0; mt < 4; ++mt)
#pragma unroll
            for (int r = 0; r < 8; ++r) {
                const int m = 16 * mt + 8 * half + r;
                const size_t b = bbase + m;
                c1r[mt][r] = c0[b * HDIM + j];
                c2r[mt][r] = c0[(size_t)BATCH * HDIM + b * HDIM + j];
                h1s[m * HDIM + j] = (__bf16)h0[b * HDIM + j];
                h2s[m * HDIM + j] = (__bf16)h0[(size_t)BATCH * HDIM + b * HDIM + j];
            }
    }
    __syncthreads();

    for (int t = 0; t < TSTEPS; ++t) {
        const bool last = (t == TSTEPS - 1);
        v8f acc[4][4];

        // ---- layer 1: gates = G0[tok] + h1 @ W_hh0^T ----
#pragma unroll
        for (int mt = 0; mt < 4; ++mt)
#pragma unroll
            for (int nt = 0; nt < 4; ++nt) acc[mt][nt] = v8f{};
        mm_acc(h1s, Wp0, wv, l15, half, acc);
        __syncthreads();   // all h1s reads done before overwrite
        lstm_update(acc, c1r, h1s, x, G0p, t, bbase, wv, l15, half,
                    last, hn, cn);
        __syncthreads();   // h1s(t) ready

        // ---- layer 2: gates = h1 @ W_ih1^T + h2 @ W_hh1^T ----
#pragma unroll
        for (int mt = 0; mt < 4; ++mt)
#pragma unroll
            for (int nt = 0; nt < 4; ++nt) acc[mt][nt] = v8f{};
        mm_acc(h1s, Wpi1, wv, l15, half, acc);
        mm_acc(h2s, Wp1,  wv, l15, half, acc);
        __syncthreads();   // all h2s reads done before overwrite
        lstm_update(acc, c2r, h2s, nullptr, G0p, t, bbase, wv, l15, half,
                    last, hn + (size_t)BATCH * HDIM, cn + (size_t)BATCH * HDIM);
        __syncthreads();   // h2s(t) ready
    }

    // ---- final FC: logits = h2(T-1) @ fc_W^T + fc_b  (20 tiles over 16 waves)
    for (int tile = wv; tile < 20; tile += NWAVES) {
        const int mt = tile / 5, nt = tile % 5;
        v8f z = v8f{};
#pragma unroll
        for (int ks = 0; ks < 8; ++ks) {
            const int k0 = 32 * ks;
            v16bf a = load_a(h2s, mt, k0, l15, half);
            v16bf b = load_b(fcp, 16 * nt + l15, k0, half);
            z = __builtin_amdgcn_wmma_f32_16x16x32_bf16(
                    false, a, false, b, (short)0, z, false, false);
        }
        const int col = 16 * nt + l15;
        if (col < VOCAB) {
            const float bias = fc_b[col];
#pragma unroll
            for (int r = 0; r < 8; ++r) {
                const int m = 16 * mt + 8 * half + r;
                logits[(size_t)(bbase + m) * VOCAB + col] = z[r] + bias;
            }
        }
    }
}

// ---------------- launcher ----------------

extern "C" void kernel_launch(void* const* d_in, const int* in_sizes, int n_in,
                              void* d_out, int out_size, void* d_ws, size_t ws_size,
                              hipStream_t stream) {
    const int*   x     = (const int*)  d_in[0];
    const float* h0    = (const float*)d_in[1];
    const float* c0    = (const float*)d_in[2];
    const float* emb   = (const float*)d_in[3];
    const float* W_ih0 = (const float*)d_in[4];
    const float* W_hh0 = (const float*)d_in[5];
    const float* W_ih1 = (const float*)d_in[6];
    const float* W_hh1 = (const float*)d_in[7];
    const float* fc_W  = (const float*)d_in[8];
    const float* fc_b  = (const float*)d_in[9];

    // workspace layout (all 16B aligned)
    char* ws = (char*)d_ws;
    float*  G0p  = (float*) (ws);                                   // 80*1024 f32
    __bf16* Wp0  = (__bf16*)(ws + 327680);                          // 512 KB
    __bf16* Wpi1 = (__bf16*)(ws + 327680 + 524288);                 // 512 KB
    __bf16* Wp1  = (__bf16*)(ws + 327680 + 2 * 524288);             // 512 KB
    __bf16* fcp  = (__bf16*)(ws + 327680 + 3 * 524288);             // 48 KB

    build_g0 <<<(VOCAB * 1024 + 255) / 256, 256, 0, stream>>>(emb, W_ih0, G0p);
    permute_w<<<(1024 * HDIM + 255) / 256, 256, 0, stream>>>(W_hh0, Wp0);
    permute_w<<<(1024 * HDIM + 255) / 256, 256, 0, stream>>>(W_ih1, Wpi1);
    permute_w<<<(1024 * HDIM + 255) / 256, 256, 0, stream>>>(W_hh1, Wp1);
    conv_fc  <<<(96 * HDIM + 255) / 256, 256, 0, stream>>>(fc_W, fcp);

    lstm_kernel<<<BATCH / MTILE, 32 * NWAVES, 0, stream>>>(
        x, h0, c0, G0p, Wp0, Wpi1, Wp1, fcp, fc_b, (float*)d_out);
}
